// GPT_63187558858984
// MI455X (gfx1250) — matrix operand order repaired and checked by reading, not compiled, not run on hardware
//
#include <hip/hip_runtime.h>
#include <hip/hip_bf16.h>

// Problem constants (match reference)
#define VSZ 50257
#define ESZ 768
#define QKS 256
#define BSZ 4
#define TSZ 1024
#define NTOK (BSZ * TSZ)     // 4096
#define LN_EPS 1e-5f

typedef __attribute__((ext_vector_type(16))) __bf16 v16bf;
typedef __attribute__((ext_vector_type(8)))  float  v8f;
typedef unsigned int v4u __attribute__((ext_vector_type(4)));
typedef int          v8i __attribute__((ext_vector_type(8)));
typedef int          v4i __attribute__((ext_vector_type(4)));

// ---------- helpers ----------

__device__ __forceinline__ unsigned short f2bf(float f) {
    unsigned int u = __float_as_uint(f);
    u += 0x7FFFu + ((u >> 16) & 1u);   // round-to-nearest-even
    return (unsigned short)(u >> 16);
}

__device__ __forceinline__ unsigned int pack_bf2(float lo, float hi) {
    unsigned int ul = __float_as_uint(lo);
    unsigned int uh = __float_as_uint(hi);
    ul += 0x7FFFu + ((ul >> 16) & 1u);
    uh += 0x7FFFu + ((uh >> 16) & 1u);
    return (ul >> 16) | (uh & 0xFFFF0000u);
}

// A-matrix 16x32 bf16 tile from row-major (ld = K stride) storage.
// Lane l (m=l&15, h=l>>4): elems 0..15 = K 16h..16h+15
__device__ __forceinline__ v16bf load_atile(const unsigned short* base, int ld, int lane) {
    int m = lane & 15, h = lane >> 4;
    const unsigned short* p = base + (size_t)m * ld + h * 16;
    union { uint4 q[2]; v16bf v; } u;
    u.q[0] = *(const uint4*)(p);
    u.q[1] = *(const uint4*)(p + 8);    // +16 bytes -> K offset +8
    return u.v;
}

// B-matrix 32x16 bf16 tile, where B[k][n] = w[n][k] and w is row-major NxK.
// Lane l (n=l&15, h=l>>4): elems 0..15 = K 16h..16h+15 (contiguous in w row n).
__device__ __forceinline__ v16bf load_btile(const unsigned short* w, int ld,
                                            int n0, int kBase, int nMax, int lane) {
    int n = n0 + (lane & 15);
    int k = kBase + ((lane >> 4) << 4);
    union { uint4 q[2]; v16bf v; } u;
    if (n < nMax) {
        const unsigned short* p = w + (size_t)n * ld + k;
        u.q[0] = *(const uint4*)(p);
        u.q[1] = *(const uint4*)(p + 8);
    } else {
        u.q[0] = make_uint4(0u, 0u, 0u, 0u);
        u.q[1] = make_uint4(0u, 0u, 0u, 0u);
    }
    return u.v;
}

// Same B tile but converting fp32 weights -> bf16 in registers (fallback path).
__device__ __forceinline__ v16bf load_btile_f32(const float* w, int ld,
                                                int n0, int kBase, int nMax, int lane) {
    int n = n0 + (lane & 15);
    int k = kBase + ((lane >> 4) << 4);
    union { unsigned int u[8]; v16bf v; } r;
    if (n < nMax) {
        const float4* p = (const float4*)(w + (size_t)n * ld + k);
        float4 f0 = p[0], f1 = p[1], f2 = p[2], f3 = p[3];
        r.u[0] = pack_bf2(f0.x, f0.y); r.u[1] = pack_bf2(f0.z, f0.w);
        r.u[2] = pack_bf2(f1.x, f1.y); r.u[3] = pack_bf2(f1.z, f1.w);
        r.u[4] = pack_bf2(f2.x, f2.y); r.u[5] = pack_bf2(f2.z, f2.w);
        r.u[6] = pack_bf2(f3.x, f3.y); r.u[7] = pack_bf2(f3.z, f3.w);
    } else {
        #pragma unroll
        for (int i = 0; i < 8; ++i) r.u[i] = 0u;
    }
    return r.v;
}

__device__ __forceinline__ v8f wmma_bf16(v16bf a, v16bf b, v8f c) {
    return __builtin_amdgcn_wmma_f32_16x16x32_bf16(false, a, false, b, (short)0, c, false, false);
}

// Issue one TDM 2D tile load: 64 rows x 32 cols of bf16 from row-major (N x 768)
// weights starting at row n0, col k, into LDS at lds_dst. Rows past nMax are
// zero-filled by the TDM's OOB handling (tensor_dim1 = nMax - n0).
// D# layout per CDNA5 ISA ch.8 (group0 128b, group1 256b; groups 2/3 zero).
__device__ __forceinline__ void tdm_issue_b(const unsigned short* lds_dst,
                                            const unsigned short* gsrc,
                                            int n0, int k, int nMax) {
    unsigned long long ga = (unsigned long long)(uintptr_t)(gsrc + (size_t)n0 * ESZ + k);
    unsigned int ldsoff = (unsigned int)(uintptr_t)lds_dst;   // low 32b = LDS offset
    int dim1 = nMax - n0;                                     // < 65536 here
    v4u g0;
    g0.x = 1u;                                                // count=1, user mode
    g0.y = ldsoff;                                            // lds_addr [63:32]
    g0.z = (unsigned int)ga;                                  // global_addr[31:0]
    g0.w = ((unsigned int)(ga >> 32) & 0x01FFFFFFu)           // global_addr[56:32]
           | 0x80000000u;                                     // type=2 ("image")
    v8i g1;
    g1[0] = 0x10000;                         // data_size=1 -> 2 bytes; no multicast
    g1[1] = (int)(0x300u << 16);             // tensor_dim0 = 768   (bits 79:48)
    g1[2] = (int)((unsigned)dim1 << 16);     // tensor_dim1 = dim1  (bits 111:80)
    g1[3] = (int)(32u << 16);                // tile_dim0 = 32      (bits 127:112)
    g1[4] = 64;                              // tile_dim1 = 64      (bits 143:128)
    g1[5] = ESZ;                             // tensor_dim0_stride = 768 (bits 207:160)
    g1[6] = 0;
    g1[7] = 0;
    v4i gz  = {0, 0, 0, 0};
    v8i gz8 = {0, 0, 0, 0, 0, 0, 0, 0};
    __builtin_amdgcn_tensor_load_to_lds(g0, g1, gz, gz, gz8, 0);
}

// ---------- kernel 1: embedding gather + layernorm ----------
__global__ void k_embed_ln(const int* __restrict__ idx, const float* __restrict__ emb,
                           const float* __restrict__ g, const float* __restrict__ beta,
                           float* __restrict__ x_out, unsigned short* __restrict__ x_bf) {
    int tok = blockIdx.x;
    int id = idx[tok];
    const float* row = emb + (size_t)id * ESZ;
    int t = threadIdx.x;
    float v0 = row[t], v1 = row[t + 256], v2 = row[t + 512];
    float s  = v0 + v1 + v2;
    float ss = v0 * v0 + v1 * v1 + v2 * v2;
    #pragma unroll
    for (int m = 16; m >= 1; m >>= 1) {
        s  += __shfl_xor(s,  m, 32);
        ss += __shfl_xor(ss, m, 32);
    }
    __shared__ float red[2][8];
    int wave = t >> 5, lane = t & 31;
    if (lane == 0) { red[0][wave] = s; red[1][wave] = ss; }
    __syncthreads();
    if (t == 0) {
        float S = 0.f, SS = 0.f;
        #pragma unroll
        for (int i = 0; i < 8; ++i) { S += red[0][i]; SS += red[1][i]; }
        float mu  = S / (float)ESZ;
        float var = SS / (float)ESZ - mu * mu;
        red[0][0] = mu;
        red[1][0] = rsqrtf(var + LN_EPS);
    }
    __syncthreads();
    float mu = red[0][0], rs = red[1][0];
    size_t base = (size_t)tok * ESZ;
    float vv[3] = {v0, v1, v2};
    #pragma unroll
    for (int i = 0; i < 3; ++i) {
        int e = t + i * 256;
        float y = (vv[i] - mu) * rs * g[e] + beta[e];
        x_out[base + e] = y;
        x_bf[base + e]  = f2bf(y);
    }
}

// ---------- kernel 2: fp32 -> bf16 weight conversion ----------
__global__ void k_cvt_w(const float* __restrict__ w, unsigned int* __restrict__ wb, long n8) {
    long i = (long)blockIdx.x * blockDim.x + threadIdx.x;   // one thread = 8 elems
    if (i >= n8) return;
    const float4* p = (const float4*)(w + i * 8);
    float4 a = p[0], b = p[1];
    uint4 o;
    o.x = pack_bf2(a.x, a.y); o.y = pack_bf2(a.z, a.w);
    o.z = pack_bf2(b.x, b.y); o.w = pack_bf2(b.z, b.w);
    ((uint4*)wb)[i] = o;
}

// ---------- kernel 3: q/k projection  out[4096, 256] = x @ w^T ----------
// One wave per 16x16 output tile; 4096 tiles; 8 waves/block -> 512 blocks.
template <bool PRECONV>
__global__ void k_proj(const unsigned short* __restrict__ xb,
                       const float* __restrict__ wf,
                       const unsigned short* __restrict__ wb,
                       unsigned short* __restrict__ outb) {
    int lane = threadIdx.x & 31;
    int lin  = blockIdx.x * 8 + (threadIdx.x >> 5);
    int m0 = (lin >> 4) << 4;        // 256 M-tiles
    int n0 = (lin & 15) << 4;        // 16 N-tiles
    v8f acc = {};
    for (int k = 0; k < ESZ; k += 32) {
        v16bf a = load_atile(xb + (size_t)m0 * ESZ + k, ESZ, lane);
        v16bf b = PRECONV ? load_btile(wb, ESZ, n0, k, QKS, lane)
                          : load_btile_f32(wf, ESZ, n0, k, QKS, lane);
        acc = wmma_bf16(a, b, acc);
    }
    int n  = n0 + (lane & 15);
    int mb = m0 + ((lane >> 4) << 3);
    #pragma unroll
    for (int j = 0; j < 8; ++j)
        outb[(size_t)(mb + j) * QKS + n] = f2bf(acc[j]);
}

// ---------- kernel 4a: logits GEMM with TDM-staged B tiles ----------
// Block = 8 waves, tile 128(M) x 64(N), K-looped 32. Wave 0 drives the Tensor
// Data Mover: double-buffered 4KB B tiles (64 x 32 bf16) into LDS, completion
// via s_wait_tensorcnt, published to the other waves with workgroup barriers.
__global__ void k_logits_tdm(const unsigned short* __restrict__ xb,
                             const unsigned short* __restrict__ wb,
                             float* __restrict__ out) {
    __shared__ unsigned short bsm[2][64 * 32];   // 2 x 4KB double buffer
    int lane = threadIdx.x & 31;
    int wv   = threadIdx.x >> 5;
    int m0   = blockIdx.y * 128 + wv * 16;
    int n0   = blockIdx.x * 64;
    v8f a0 = {}, a1 = {}, a2 = {}, a3 = {};
    const int KSTEPS = ESZ / 32;

    if (wv == 0) tdm_issue_b(&bsm[0][0], wb, n0, 0, VSZ);

    for (int i = 0; i < KSTEPS; ++i) {
        __syncthreads();                 // buf[(i+1)&1] no longer read by anyone
        if (wv == 0) {
            if (i + 1 < KSTEPS) {
                tdm_issue_b(&bsm[(i + 1) & 1][0], wb, n0, (i + 1) * 32, VSZ);
                __builtin_amdgcn_s_wait_tensorcnt(1);   // tile i complete
            } else {
                __builtin_amdgcn_s_wait_tensorcnt(0);
            }
        }
        __syncthreads();                 // tile i visible to all waves
        int k = i * 32;
        v16bf a = load_atile(xb + (size_t)m0 * ESZ + k, ESZ, lane);
        const unsigned short* bs = &bsm[i & 1][0];
        // LDS tile is 64 rows x 32 bf16, row stride 32 elements.
        int kl = (lane >> 4) << 4;
        int nl = lane & 15;
        union { uint4 q[2]; v16bf v; } u0, u1, u2, u3;
        u0.q[0] = *(const uint4*)(bs + (nl +  0) * 32 + kl);
        u0.q[1] = *(const uint4*)(bs + (nl +  0) * 32 + kl + 8);
        u1.q[0] = *(const uint4*)(bs + (nl + 16) * 32 + kl);
        u1.q[1] = *(const uint4*)(bs + (nl + 16) * 32 + kl + 8);
        u2.q[0] = *(const uint4*)(bs + (nl + 32) * 32 + kl);
        u2.q[1] = *(const uint4*)(bs + (nl + 32) * 32 + kl + 8);
        u3.q[0] = *(const uint4*)(bs + (nl + 48) * 32 + kl);
        u3.q[1] = *(const uint4*)(bs + (nl + 48) * 32 + kl + 8);
        a0 = wmma_bf16(a, u0.v, a0);
        a1 = wmma_bf16(a, u1.v, a1);
        a2 = wmma_bf16(a, u2.v, a2);
        a3 = wmma_bf16(a, u3.v, a3);
    }
    int n  = lane & 15;
    int mb = m0 + ((lane >> 4) << 3);
    #pragma unroll
    for (int j = 0; j < 8; ++j) {
        size_t rb = (size_t)(mb + j) * VSZ;
        int c0 = n0 + n;
        if (c0 < VSZ)       out[rb + c0]      = a0[j];
        if (c0 + 16 < VSZ)  out[rb + c0 + 16] = a1[j];
        if (c0 + 32 < VSZ)  out[rb + c0 + 32] = a2[j];
        if (c0 + 48 < VSZ)  out[rb + c0 + 48] = a3[j];
    }
}

// ---------- kernel 4b: logits GEMM fallback (no preconverted weights) ----------
__global__ void k_logits_fb(const unsigned short* __restrict__ xb,
                            const float* __restrict__ wf,
                            float* __restrict__ out) {
    int lane = threadIdx.x & 31;
    int wv   = threadIdx.x >> 5;
    int m0   = blockIdx.y * 128 + wv * 16;
    int n0   = blockIdx.x * 64;
    v8f a0 = {}, a1 = {}, a2 = {}, a3 = {};
    for (int k = 0; k < ESZ; k += 32) {
        v16bf a = load_atile(xb + (size_t)m0 * ESZ + k, ESZ, lane);
        v16bf b0 = load_btile_f32(wf, ESZ, n0,      k, VSZ, lane);
        v16bf b1 = load_btile_f32(wf, ESZ, n0 + 16, k, VSZ, lane);
        v16bf b2 = load_btile_f32(wf, ESZ, n0 + 32, k, VSZ, lane);
        v16bf b3 = load_btile_f32(wf, ESZ, n0 + 48, k, VSZ, lane);
        a0 = wmma_bf16(a, b0, a0);
        a1 = wmma_bf16(a, b1, a1);
        a2 = wmma_bf16(a, b2, a2);
        a3 = wmma_bf16(a, b3, a3);
    }
    int n  = lane & 15;
    int mb = m0 + ((lane >> 4) << 3);
    #pragma unroll
    for (int j = 0; j < 8; ++j) {
        size_t rb = (size_t)(mb + j) * VSZ;
        int c0 = n0 + n;
        if (c0 < VSZ)       out[rb + c0]      = a0[j];
        if (c0 + 16 < VSZ)  out[rb + c0 + 16] = a1[j];
        if (c0 + 32 < VSZ)  out[rb + c0 + 32] = a2[j];
        if (c0 + 48 < VSZ)  out[rb + c0 + 48] = a3[j];
    }
}

// ---------- kernel 5: causal c = q k^T / QK, scattered into logits ----------
// grid (T/16 s-tiles, T/16 t-tiles, B); one wave per tile; upper tiles exit
// whole-wave so EXEC is all-ones at every WMMA.
__global__ void k_copy(const unsigned short* __restrict__ qb,
                       const unsigned short* __restrict__ kb,
                       const int* __restrict__ idx, float* __restrict__ out) {
    int st = blockIdx.x, tt = blockIdx.y, b = blockIdx.z;
    if (st > tt) return;
    int lane = threadIdx.x;
    int t0 = tt * 16, s0 = st * 16;
    const unsigned short* qbase = qb + ((size_t)(b * TSZ + t0)) * QKS;
    const unsigned short* kbase = kb + ((size_t)(b * TSZ + s0)) * QKS;
    v8f acc = {};
    #pragma unroll
    for (int k = 0; k < QKS; k += 32) {
        v16bf a  = load_atile(qbase + k, QKS, lane);
        v16bf bm = load_btile(kbase, QKS, 0, k, 1 << 30, lane);
        acc = wmma_bf16(a, bm, acc);
    }
    int s   = s0 + (lane & 15);
    int vid = idx[b * TSZ + s];          // one column per lane -> one idx load
    int tb  = t0 + ((lane >> 4) << 3);
    const float scale = 1.0f / (float)QKS;
    #pragma unroll
    for (int j = 0; j < 8; ++j) {
        int t = tb + j;
        if (s <= t)
            atomicAdd(out + ((size_t)(b * TSZ + t)) * VSZ + vid, acc[j] * scale);
    }
}

// ---------- launch ----------
extern "C" void kernel_launch(void* const* d_in, const int* in_sizes, int n_in,
                              void* d_out, int out_size, void* d_ws, size_t ws_size,
                              hipStream_t stream) {
    const int*   idx    = (const int*)d_in[0];
    const float* emb_w  = (const float*)d_in[1];
    const float* ln_g   = (const float*)d_in[2];
    const float* ln_b   = (const float*)d_in[3];
    const float* head_w = (const float*)d_in[4];
    const float* hq_w   = (const float*)d_in[5];
    const float* hk_w   = (const float*)d_in[6];

    float* logits = (float*)d_out;                       // (B,T,V)
    float* x_out  = logits + (size_t)NTOK * VSZ;         // (B,T,E), tuple output #2

    // workspace: x_bf16 | q_bf16 | k_bf16 | hq_bf16 | hk_bf16 | head_w bf16
    unsigned short* xb  = (unsigned short*)d_ws;                 //  6,291,456 B
    unsigned short* qb  = xb  + (size_t)NTOK * ESZ;              //  2,097,152 B
    unsigned short* kb  = qb  + (size_t)NTOK * QKS;              //  2,097,152 B
    unsigned short* hqb = kb  + (size_t)NTOK * QKS;              //    393,216 B
    unsigned short* hkb = hqb + (size_t)QKS * ESZ;               //    393,216 B
    unsigned short* wb  = hkb + (size_t)QKS * ESZ;               // 77,194,752 B
    const size_t need = (size_t)NTOK * ESZ * 2 + (size_t)NTOK * QKS * 4
                      + (size_t)QKS * ESZ * 4 + (size_t)VSZ * ESZ * 2;
    bool preconv = (ws_size >= need);

    // 1) gather + layernorm
    k_embed_ln<<<NTOK, 256, 0, stream>>>(idx, emb_w, ln_g, ln_b, x_out, xb);

    if (preconv) {
        // 2) fp32 -> bf16 weight conversions
        long nqk = (long)QKS * ESZ / 8;                  // 24576
        k_cvt_w<<<(int)((nqk + 255) / 256), 256, 0, stream>>>(hq_w, (unsigned int*)hqb, nqk);
        k_cvt_w<<<(int)((nqk + 255) / 256), 256, 0, stream>>>(hk_w, (unsigned int*)hkb, nqk);
        long nw = (long)VSZ * ESZ / 8;                   // 4,824,672
        k_cvt_w<<<(int)((nw + 255) / 256), 256, 0, stream>>>(head_w, (unsigned int*)wb, nw);

        // 3) q / k projections (pure bf16 loads + WMMA)
        k_proj<true ><<<512, 256, 0, stream>>>(xb, hq_w, hqb, qb);
        k_proj<true ><<<512, 256, 0, stream>>>(xb, hk_w, hkb, kb);

        // 4) big GEMM with TDM-staged B tiles
        dim3 g3((VSZ + 63) / 64, NTOK / 128);
        k_logits_tdm<<<g3, 256, 0, stream>>>(xb, wb, logits);
    } else {
        k_proj<false><<<512, 256, 0, stream>>>(xb, hq_w, nullptr, qb);
        k_proj<false><<<512, 256, 0, stream>>>(xb, hk_w, nullptr, kb);
        dim3 g3((VSZ + 63) / 64, NTOK / 128);
        k_logits_fb<<<g3, 256, 0, stream>>>(xb, head_w, logits);
    }

    // 5) causal QK copy scatter into logits
    dim3 g5(TSZ / 16, TSZ / 16, BSZ);
    k_copy<<<g5, 32, 0, stream>>>(qb, kb, idx, logits);
}
